// AFS_12549894439384
// MI455X (gfx1250) — matrix operand order repaired and checked by reading, not compiled, not run on hardware
//
#include <hip/hip_runtime.h>

typedef __attribute__((ext_vector_type(16))) __bf16 v16bf;
typedef __attribute__((ext_vector_type(8)))  float  v8f;
typedef unsigned short u16;
typedef unsigned int   u32;
typedef unsigned long long u64;

#define B_DIM   8192
#define F_DIMC  2048
#define E_NODE  32
#define HIDDEN  2048
#define CLASSES 1000
#define NPAD    1024
#define NATT    4096   // F_DIM * 2 (f,a) flattened

#if __has_builtin(__builtin_amdgcn_cvt_pk_bf16_f32)
#define HAVE_PKBF 1
#else
#define HAVE_PKBF 0
#endif

#if __has_builtin(__builtin_amdgcn_tensor_load_to_lds) && __has_builtin(__builtin_amdgcn_s_wait_tensorcnt)
#define HAVE_TDM 1
#else
#define HAVE_TDM 0
#endif

// ---------- scalar/pack helpers ----------
__device__ __forceinline__ u32 rne_hi(float f) {   // round f to bf16, keep in [31:16]
  u32 u = __float_as_uint(f);
  return u + (0x7FFFu + ((u >> 16) & 1u));
}
__device__ __forceinline__ u16 f2bf(float f) {
#if HAVE_PKBF
  auto r = __builtin_amdgcn_cvt_pk_bf16_f32(f, 0.0f);
  return (u16)__builtin_bit_cast(u32, r);
#else
  return (u16)(rne_hi(f) >> 16);
#endif
}
__device__ __forceinline__ u32 pk2(float a, float b) {
#if HAVE_PKBF
  auto r = __builtin_amdgcn_cvt_pk_bf16_f32(a, b);
  return __builtin_bit_cast(u32, r);
#else
  // single v_perm_b32 merges the two rounded high halves: {b.hi, a.hi}
  return __builtin_amdgcn_perm(rne_hi(b), rne_hi(a), 0x07060302u);
#endif
}
__device__ __forceinline__ float fast_tanh(float x) {
#if __has_builtin(__builtin_amdgcn_tanhf)
  return __builtin_amdgcn_tanhf(x);
#elif __has_builtin(__builtin_amdgcn_tanh_f32)
  return __builtin_amdgcn_tanh_f32(x);
#else
  // CDNA5 ISA lists TANH as a 32-bit transcendental VALU op (S_DELAY_ALU sec.)
  float y;
  asm("v_tanh_f32 %0, %1" : "=v"(y) : "v"(x));
  return y;
#endif
}

union Frag { uint4 u[2]; v16bf v; };

__device__ __forceinline__ v8f wmma_bf16(const Frag& a, const Frag& b, v8f c) {
  return __builtin_amdgcn_wmma_f32_16x16x32_bf16(false, a.v, false, b.v,
                                                 (short)0, c, false, false);
}

// A-matrix fragment gather from LDS tile stored row-major [row][32] bf16.
// lanes 0-15: K = {0..7, 16..23}; lanes 16-31: +8  (ISA 7.12.2, 16-bit A 16x32)
__device__ __forceinline__ Frag loadA(const u16* ldsRow, int hi) {
  const uint4* p = (const uint4*)ldsRow;
  Frag f; f.u[0] = p[hi]; f.u[1] = p[2 + hi];
  return f;
}
// B-matrix fragment gather from LDS tile stored TRANSPOSED [n][32] bf16.
// lanes 0-15: K = {0..7, 16..23}; lanes 16-31: +8 (same K order as A)
__device__ __forceinline__ Frag loadB(const u16* ldsRow, int hi) {
  const uint4* p = (const uint4*)ldsRow;
  Frag f; f.u[0] = p[hi]; f.u[1] = p[2 + hi];
  return f;
}

// ---------- Tensor Data Mover: 2D bf16 tile -> LDS (D# per ISA ch.8) ----------
#if HAVE_TDM
typedef __attribute__((ext_vector_type(4))) unsigned int v4u_t;
typedef __attribute__((ext_vector_type(8))) int v8i_t;
typedef __attribute__((ext_vector_type(4))) int v4i_t;

__device__ __forceinline__ void tdm_load_2d_bf16(u32 lds_off, const u16* gptr,
                                                 u32 tile0, u32 tile1, u64 stride0) {
  const u64 ga = (u64)(const void*)gptr;
  const u32 td0 = 0x00100000u;   // generous tensor dims: tiles never clip
  const u32 td1 = 0x00100000u;
  v4u_t g0;
  g0[0] = 1u;                                              // count=1 (valid user D#)
  g0[1] = lds_off;                                         // lds_addr (bytes)
  g0[2] = (u32)ga;                                         // global_addr[31:0]
  g0[3] = ((u32)(ga >> 32) & 0x01FFFFFFu) | 0x80000000u;   // addr[56:32] | type=2
  v8i_t g1;
  g1[0] = (int)0x00010000u;                                // data_size=1 (2B); no multicast
  g1[1] = (int)((td0 & 0xFFFFu) << 16);                    // tensor_dim0[15:0]
  g1[2] = (int)((td0 >> 16) | ((td1 & 0xFFFFu) << 16));    // td0[31:16] | td1[15:0]
  g1[3] = (int)((td1 >> 16) | (tile0 << 16));              // td1[31:16] | tile_dim0
  g1[4] = (int)(tile1 & 0xFFFFu);                          // tile_dim1 (tile_dim2=0)
  g1[5] = (int)(u32)stride0;                               // tensor_dim0_stride[31:0]
  g1[6] = (int)(u32)((stride0 >> 32) & 0xFFFFu);           // stride0[47:32] | stride1=0
  g1[7] = 0;
  v4i_t z4 = {0, 0, 0, 0};
  v8i_t z8 = {0, 0, 0, 0, 0, 0, 0, 0};
  // 6-arg toolchain form: (g0, g1, g2, g3, g4, cpol)
  __builtin_amdgcn_tensor_load_to_lds(g0, g1, z4, z4, z8, 0);
}
__device__ __forceinline__ u32 lds_off(const void* p) {
  return (u32)(size_t)p;   // flat LDS address: [31:0] is the in-allocation offset
}
#endif

// ---------- conversion / transpose kernels (one-time, small) ----------
__global__ void k_cvt_We(const float* __restrict__ We, u16* __restrict__ Wet) {
  int i = blockIdx.x * 256 + threadIdx.x;          // Wet[n][k] n<32 k<2048
  if (i >= E_NODE * F_DIMC) return;
  int n = i >> 11, k = i & 2047;
  Wet[i] = f2bf(We[k * E_NODE + n]);
}
__global__ void k_cvt_Wa(const float* __restrict__ Wa, u16* __restrict__ Wat) {
  int i = blockIdx.x * 256 + threadIdx.x;          // Wat[n][e] n=f*2+a
  if (i >= NATT * E_NODE) return;
  int n = i >> 5, e = i & 31, f = n >> 1, a = n & 1;
  Wat[i] = f2bf(Wa[f * 64 + e * 2 + a]);
}
__global__ void k_cvt_W1(const float* __restrict__ W1, u16* __restrict__ W1t) {
  int i = blockIdx.x * 256 + threadIdx.x;          // coalesced read, scatter write
  if (i >= F_DIMC * HIDDEN) return;
  int k = i >> 11, n = i & 2047;                   // i = k*2048+n
  W1t[(size_t)n * F_DIMC + k] = f2bf(W1[i]);
}
__global__ void k_cvt_W2(const float* __restrict__ W2, u16* __restrict__ W2t) {
  int i = blockIdx.x * 256 + threadIdx.x;          // W2t[1024][2048], zero-padded
  if (i >= NPAD * HIDDEN) return;
  int n = i >> 11, k = i & 2047;
  W2t[i] = (n < CLASSES) ? f2bf(W2[(size_t)k * CLASSES + n]) : (u16)0;
}

// ---------- K1: e = tanh(x @ We + be) -> bf16 [8192][32] ----------
__global__ __launch_bounds__(256) void k_egemm(const float* __restrict__ x,
                                               const float* __restrict__ be,
                                               const u16* __restrict__ Wet,
                                               u16* __restrict__ e_bf) {
  __shared__ __align__(16) u16 sA[128][32];
  __shared__ __align__(16) u16 sB[32][32];        // transposed [n][k]
  const int tid = threadIdx.x;
  const int lane = tid & 31, wave = tid >> 5;
  const int laneLo = lane & 15, hi = (lane >> 4) & 1;
  const int m0 = blockIdx.x * 128;
  const int rowA = tid >> 1, csA = (tid & 1) * 16;

  v8f acc0 = {}, acc1 = {};
  for (int kt = 0; kt < F_DIMC; kt += 32) {
    __syncthreads();
    { // A: 128x32 f32 -> bf16 (rne + v_perm pack)
      const float4* xp = (const float4*)&x[(size_t)(m0 + rowA) * F_DIMC + kt + csA];
      float4 f0 = xp[0], f1 = xp[1], f2 = xp[2], f3 = xp[3];
      uint4 q0, q1;
      q0.x = pk2(f0.x, f0.y); q0.y = pk2(f0.z, f0.w);
      q0.z = pk2(f1.x, f1.y); q0.w = pk2(f1.z, f1.w);
      q1.x = pk2(f2.x, f2.y); q1.y = pk2(f2.z, f2.w);
      q1.z = pk2(f3.x, f3.y); q1.w = pk2(f3.z, f3.w);
      uint4* d = (uint4*)&sA[rowA][csA];
      d[0] = q0; d[1] = q1;
    }
    if (tid < 128) { // B: Wet rows 0..31, cols kt..kt+31
      int n = tid >> 2, c = (tid & 3) * 8;
      *(uint4*)&sB[n][c] = *(const uint4*)&Wet[(size_t)n * F_DIMC + kt + c];
    }
    if (kt + 32 < F_DIMC)
      __builtin_prefetch(&x[(size_t)(m0 + rowA) * F_DIMC + kt + 32 + csA], 0, 0);
    __syncthreads();
    Frag a = loadA(&sA[wave * 16 + laneLo][0], hi);
    Frag b0 = loadB(&sB[laneLo][0], hi);
    Frag b1 = loadB(&sB[16 + laneLo][0], hi);
    acc0 = wmma_bf16(a, b0, acc0);
    acc1 = wmma_bf16(a, b1, acc1);
  }
#pragma unroll
  for (int r = 0; r < 8; ++r) {
    int m = m0 + wave * 16 + r + hi * 8;
    e_bf[(size_t)m * 32 + laneLo]      = f2bf(fast_tanh(acc0[r] + be[laneLo]));
    e_bf[(size_t)m * 32 + 16 + laneLo] = f2bf(fast_tanh(acc1[r] + be[16 + laneLo]));
  }
}

// ---------- K2: attention + weighted (fused gate GEMM, K=32) ----------
__global__ __launch_bounds__(256) void k_att(const u16* __restrict__ e_bf,
                                             const u16* __restrict__ Wat,
                                             const float* __restrict__ ba,
                                             const float* __restrict__ x,
                                             float* __restrict__ att_out,
                                             u16* __restrict__ wgt) {
  __shared__ __align__(16) u16 sE[128][32];       // A tile (b x e)
  __shared__ __align__(16) u16 sW[256][32];       // B tile transposed [n][e]
  const int tid = threadIdx.x;
  const int lane = tid & 31, wave = tid >> 5;
  const int laneLo = lane & 15, hi = (lane >> 4) & 1;
  const int m0 = blockIdx.y * 128, n0 = blockIdx.x * 256;

#if HAVE_TDM
  if (wave == 0) {   // both tiles are contiguous in global: 1-row TDM descriptors
    tdm_load_2d_bf16(lds_off(&sE[0][0]), e_bf + (size_t)m0 * 32, 4096u, 1u, 4096u);
    tdm_load_2d_bf16(lds_off(&sW[0][0]), Wat + (size_t)n0 * 32, 8192u, 1u, 8192u);
    __builtin_amdgcn_s_wait_tensorcnt(0);
  }
#else
  for (int i = tid; i < 128 * 4; i += 256) {
    int row = i >> 2, c = (i & 3) * 8;
    *(uint4*)&sE[row][c] = *(const uint4*)&e_bf[(size_t)(m0 + row) * 32 + c];
  }
  for (int i = tid; i < 256 * 4; i += 256) {
    int row = i >> 2, c = (i & 3) * 8;
    *(uint4*)&sW[row][c] = *(const uint4*)&Wat[(size_t)(n0 + row) * 32 + c];
  }
#endif
  __syncthreads();

  Frag a = loadA(&sE[wave * 16 + laneLo][0], hi); // wave owns one 16-row m-tile
  for (int nt = 0; nt < 16; ++nt) {
    Frag b = loadB(&sW[nt * 16 + laneLo][0], hi);
    v8f c = {};
    c = wmma_bf16(a, b, c);
    const int nG = n0 + nt * 16 + laneLo;         // global n = f*2+a
    const float bav = ba[nG];
    const int f = nG >> 1;
    const bool a0 = (laneLo & 1) == 0;
#pragma unroll
    for (int r = 0; r < 8; ++r) {
      float s = fast_tanh(c[r] + bav);
      float sp = __shfl_xor(s, 1, 32);            // partner within (a0,a1) pair
      if (a0) {                                   // softmax over 2 -> sigmoid(s1-s0)
        float attv = 1.0f / (1.0f + __expf(s - sp));
        int m = m0 + wave * 16 + r + hi * 8;
        att_out[(size_t)m * F_DIMC + f] = attv;
        wgt[(size_t)m * F_DIMC + f] = f2bf(x[(size_t)m * F_DIMC + f] * attv);
      }
    }
  }
}

// ---------- K3/K4 tiled GEMM: C[128][128] per block, TDM double-buffered ----------
// waves: 4 along M x 2 along N; per wave 2 m-tiles x 4 n-tiles
template <bool RELU_BF16_OUT>
__global__ __launch_bounds__(256) void k_gemm(const u16* __restrict__ A,
                                              const u16* __restrict__ Bt,
                                              const float* __restrict__ bias,
                                              u16* __restrict__ outBf,
                                              float* __restrict__ outF,
                                              int Nstore) {
  __shared__ __align__(16) u16 sA[2][128][32];
  __shared__ __align__(16) u16 sB[2][128][32];    // transposed [n][k]
  const int tid = threadIdx.x;
  const int lane = tid & 31, wave = tid >> 5;
  const int laneLo = lane & 15, hi = (lane >> 4) & 1;
  const int wm = wave >> 1, wn = wave & 1;
  const int m0 = blockIdx.y * 128, n0 = blockIdx.x * 128;

  v8f acc[2][4];
  v8f zero = {};
#pragma unroll
  for (int i = 0; i < 2; ++i)
#pragma unroll
    for (int j = 0; j < 4; ++j) acc[i][j] = zero;

  const int NK = 2048 / 32;   // 64 k-steps

#if HAVE_TDM
  const u32 offA[2] = { lds_off(&sA[0][0][0]), lds_off(&sA[1][0][0]) };
  const u32 offB[2] = { lds_off(&sB[0][0][0]), lds_off(&sB[1][0][0]) };
  if (wave == 0) {   // prologue: DMA tile 0 into buffer 0
    tdm_load_2d_bf16(offA[0], A  + (size_t)m0 * 2048, 32u, 128u, 2048u);
    tdm_load_2d_bf16(offB[0], Bt + (size_t)n0 * 2048, 32u, 128u, 2048u);
  }
  for (int i = 0; i < NK; ++i) {
    const int p = i & 1;
    if (wave == 0) {
      if (i + 1 < NK) {   // software pipeline: DMA next tile into other buffer
        tdm_load_2d_bf16(offA[1 - p], A  + (size_t)m0 * 2048 + (size_t)(i + 1) * 32,
                         32u, 128u, 2048u);
        tdm_load_2d_bf16(offB[1 - p], Bt + (size_t)n0 * 2048 + (size_t)(i + 1) * 32,
                         32u, 128u, 2048u);
        __builtin_amdgcn_s_wait_tensorcnt(2);   // tile i done (in-order), i+1 in flight
      } else {
        __builtin_amdgcn_s_wait_tensorcnt(0);
      }
    }
    __syncthreads();   // tile i visible to all waves
    Frag af[2], bfr[4];
#pragma unroll
    for (int mt = 0; mt < 2; ++mt)
      af[mt] = loadA(&sA[p][wm * 32 + mt * 16 + laneLo][0], hi);
#pragma unroll
    for (int nt = 0; nt < 4; ++nt)
      bfr[nt] = loadB(&sB[p][wn * 64 + nt * 16 + laneLo][0], hi);
#pragma unroll
    for (int mt = 0; mt < 2; ++mt)
#pragma unroll
      for (int nt = 0; nt < 4; ++nt)
        acc[mt][nt] = wmma_bf16(af[mt], bfr[nt], acc[mt][nt]);
    __syncthreads();   // all reads of buffer p done before it is DMA-refilled
  }
#else
  const int row = tid >> 1, cs = (tid & 1) * 16;
  for (int i = 0; i < NK; ++i) {
    const int kt = i * 32;
    __syncthreads();
    {
      const uint4* s = (const uint4*)&A[(size_t)(m0 + row) * 2048 + kt + cs];
      uint4* d = (uint4*)&sA[0][row][cs];
      d[0] = s[0]; d[1] = s[1];
    }
    {
      const uint4* s = (const uint4*)&Bt[(size_t)(n0 + row) * 2048 + kt + cs];
      uint4* d = (uint4*)&sB[0][row][cs];
      d[0] = s[0]; d[1] = s[1];
    }
    if (kt + 32 < 2048) {
      __builtin_prefetch(&A[(size_t)(m0 + row) * 2048 + kt + 32 + cs], 0, 0);
      __builtin_prefetch(&Bt[(size_t)(n0 + row) * 2048 + kt + 32 + cs], 0, 0);
    }
    __syncthreads();
    Frag af[2], bfr[4];
#pragma unroll
    for (int mt = 0; mt < 2; ++mt)
      af[mt] = loadA(&sA[0][wm * 32 + mt * 16 + laneLo][0], hi);
#pragma unroll
    for (int nt = 0; nt < 4; ++nt)
      bfr[nt] = loadB(&sB[0][wn * 64 + nt * 16 + laneLo][0], hi);
#pragma unroll
    for (int mt = 0; mt < 2; ++mt)
#pragma unroll
      for (int nt = 0; nt < 4; ++nt)
        acc[mt][nt] = wmma_bf16(af[mt], bfr[nt], acc[mt][nt]);
  }
#endif

#pragma unroll
  for (int mt = 0; mt < 2; ++mt) {
#pragma unroll
    for (int nt = 0; nt < 4; ++nt) {
      const int n = n0 + wn * 64 + nt * 16 + laneLo;
      const float bv = (n < Nstore) ? bias[n] : 0.0f;
#pragma unroll
      for (int r = 0; r < 8; ++r) {
        int m = m0 + wm * 32 + mt * 16 + r + hi * 8;
        float v = acc[mt][nt][r] + bv;
        if (RELU_BF16_OUT) {
          v = fmaxf(v, 0.0f);
          outBf[(size_t)m * 2048 + n] = f2bf(v);
        } else if (n < Nstore) {
          outF[(size_t)m * Nstore + n] = v;
        }
      }
    }
  }
}

// ---------- host launch ----------
extern "C" void kernel_launch(void* const* d_in, const int* in_sizes, int n_in,
                              void* d_out, int out_size, void* d_ws, size_t ws_size,
                              hipStream_t stream) {
  (void)in_sizes; (void)n_in; (void)out_size; (void)ws_size;
  const float* x  = (const float*)d_in[0];
  const float* We = (const float*)d_in[1];
  const float* be = (const float*)d_in[2];
  const float* Wa = (const float*)d_in[3];
  const float* ba = (const float*)d_in[4];
  const float* W1 = (const float*)d_in[5];
  const float* b1 = (const float*)d_in[6];
  const float* W2 = (const float*)d_in[7];
  const float* b2 = (const float*)d_in[8];

  float* out = (float*)d_out;                       // [8192][1000]
  float* att = out + (size_t)B_DIM * CLASSES;       // [8192][2048]

  char* ws = (char*)d_ws;
  size_t o = 0;
  u16* Wet  = (u16*)(ws + o); o += (size_t)E_NODE * F_DIMC * 2;   // 128 KB
  u16* Wat  = (u16*)(ws + o); o += (size_t)NATT * E_NODE * 2;     // 256 KB
  u16* W1t  = (u16*)(ws + o); o += (size_t)HIDDEN * F_DIMC * 2;   // 8 MB
  u16* W2t  = (u16*)(ws + o); o += (size_t)NPAD * HIDDEN * 2;     // 4 MB
  u16* e_bf = (u16*)(ws + o); o += (size_t)B_DIM * E_NODE * 2;    // 512 KB
  u16* wgt  = (u16*)(ws + o); o += (size_t)B_DIM * F_DIMC * 2;    // 32 MB
  u16* hbf  = (u16*)(ws + o); o += (size_t)B_DIM * HIDDEN * 2;    // 32 MB

  hipLaunchKernelGGL(k_cvt_We, dim3((E_NODE * F_DIMC + 255) / 256), dim3(256), 0, stream, We, Wet);
  hipLaunchKernelGGL(k_cvt_Wa, dim3((NATT * E_NODE + 255) / 256), dim3(256), 0, stream, Wa, Wat);
  hipLaunchKernelGGL(k_cvt_W1, dim3((F_DIMC * HIDDEN + 255) / 256), dim3(256), 0, stream, W1, W1t);
  hipLaunchKernelGGL(k_cvt_W2, dim3((NPAD * HIDDEN + 255) / 256), dim3(256), 0, stream, W2, W2t);

  hipLaunchKernelGGL(k_egemm, dim3(B_DIM / 128), dim3(256), 0, stream, x, be, Wet, e_bf);
  hipLaunchKernelGGL(k_att, dim3(NATT / 256, B_DIM / 128), dim3(256), 0, stream,
                     e_bf, Wat, ba, x, att, wgt);
  hipLaunchKernelGGL((k_gemm<true>), dim3(HIDDEN / 128, B_DIM / 128), dim3(256), 0, stream,
                     wgt, W1t, b1, hbf, (float*)nullptr, HIDDEN);
  hipLaunchKernelGGL((k_gemm<false>), dim3(NPAD / 128, B_DIM / 128), dim3(256), 0, stream,
                     hbf, W2t, b2, (u16*)nullptr, out, CLASSES);
}
